// LocalMHAttention_53850299957571
// MI455X (gfx1250) — matrix operand order repaired and checked by reading, not compiled, not run on hardware
//
#include <hip/hip_runtime.h>
#include <math.h>

// ---------------------------------------------------------------------------
// LocalMHAttention for MI455X (gfx1250, wave32, WMMA).
// Attention QK / PV matmuls run on v_wmma_f32_16x16x32_f16 (f32 accumulate).
// Each wave processes TWO 16-query tiles so K/V operand loads are amortized.
// Workspace usage: ~46 MB.
// ---------------------------------------------------------------------------

typedef __attribute__((ext_vector_type(16))) _Float16 v16h;
typedef __attribute__((ext_vector_type(8)))  float    v8f;

#define BSZ   8
#define CCH   256
#define NN    1500
#define HH    8
#define DKK   32
#define NT16  94            // ceil(N/16)
#define NC32  47            // ceil(N/32)
#define NPAD  (NT16 * 16)   // 1504
#define NWRD  47            // mask words per row (covers m up to 1503)
#define IOU_THR 0.5f
#define EPSV  1e-8f

// ---- wave32 cross-lane helpers via raw CDNA5 builtins ---------------------
__device__ __forceinline__ float swap16f(float x) {        // ds_swizzle SWAPX16
  return __int_as_float(__builtin_amdgcn_ds_swizzle(__float_as_int(x), 0x401F));
}
__device__ __forceinline__ int swap16i(int x) {
  return __builtin_amdgcn_ds_swizzle(x, 0x401F);
}
__device__ __forceinline__ float bcastf(float x, int srcLane) { // ds_bpermute
  return __int_as_float(__builtin_amdgcn_ds_bpermute(srcLane << 2, __float_as_int(x)));
}
__device__ __forceinline__ int bcasti(int x, int srcLane) {
  return __builtin_amdgcn_ds_bpermute(srcLane << 2, x);
}

__device__ __forceinline__ v8f wmma_f16(v16h a, v16h b, v8f c) {
  return __builtin_amdgcn_wmma_f32_16x16x32_f16(false, a, false, b, (short)0, c,
                                                false, false);
}

// ---------------------------------------------------------------------------
// Kernel 1: transpose w_proj (256x256) -> wT[j][i] = w[i][j]
// ---------------------------------------------------------------------------
__global__ void k_transpose_w(const float* __restrict__ w, float* __restrict__ wT) {
  int idx = blockIdx.x * 256 + threadIdx.x;   // idx = j*256 + i
  int j = idx >> 8, i = idx & 255;
  wT[idx] = w[i * 256 + j];
}

// ---------------------------------------------------------------------------
// Kernel 2: head projections q/k/v = W(32x32) @ x(32xN) per (b,h).
// Emits:
//   qswz : f16, WMMA B-layout (32d x 16n) per 16-n tile
//   kswz : f16, WMMA A-layout (16m x 32d) per 16-m tile
//   vswz : f16, WMMA B-layout (32m x 16d) per 32-m chunk, 2 d-groups
//   vf32 : f32 plain [bh][d][n] (for the no-neighbor fallback)
// Padded n in [N, 1504) are written as zeros (NaN-free WMMA inputs).
// ---------------------------------------------------------------------------
__global__ void k_head_proj(const float* __restrict__ q_in,
                            const float* __restrict__ k_in,
                            const float* __restrict__ v_in,
                            const float* __restrict__ wq,
                            const float* __restrict__ wk,
                            const float* __restrict__ wv,
                            _Float16* __restrict__ qswz,
                            _Float16* __restrict__ kswz,
                            _Float16* __restrict__ vswz,
                            float* __restrict__ vf32) {
  int bh    = blockIdx.x / 12;
  int chunk = blockIdx.x % 12;
  int n     = chunk * 128 + threadIdx.x;
  if (n >= NPAD) return;
  int b = bh >> 3, h = bh & 7;
  bool valid = (n < NN);
  const size_t xbase = ((size_t)b * CCH + (size_t)h * DKK) * NN + n;

  float x[DKK], y[DKK];

  // ---- Q: y = wq[h] @ x, store B-layout ----
  #pragma unroll
  for (int j = 0; j < DKK; j++) x[j] = valid ? q_in[xbase + (size_t)j * NN] : 0.0f;
  for (int i = 0; i < DKK; i++) {
    const float* wr = wq + (h * DKK + i) * DKK;
    float a = 0.0f;
    #pragma unroll
    for (int j = 0; j < DKK; j++) a = fmaf(wr[j], x[j], a);
    y[i] = a;
  }
  {
    _Float16* base = qswz + ((size_t)bh * NT16 + (n >> 4)) * 512;
    int ncol = n & 15;
    #pragma unroll
    for (int d = 0; d < DKK; d++) {
      int lane = ncol + (((d >> 3) & 1) << 4);
      int half = (d & 7) + ((d >> 4) << 3);
      base[lane * 16 + half] = (_Float16)y[d];
    }
  }

  // ---- K: A-layout ----
  #pragma unroll
  for (int j = 0; j < DKK; j++) x[j] = valid ? k_in[xbase + (size_t)j * NN] : 0.0f;
  for (int i = 0; i < DKK; i++) {
    const float* wr = wk + (h * DKK + i) * DKK;
    float a = 0.0f;
    #pragma unroll
    for (int j = 0; j < DKK; j++) a = fmaf(wr[j], x[j], a);
    y[i] = a;
  }
  {
    _Float16* base = kswz + ((size_t)bh * NT16 + (n >> 4)) * 512;
    int mloc = n & 15;
    #pragma unroll
    for (int d = 0; d < DKK; d++) {
      int lane = mloc + (((d >> 3) & 1) << 4);
      int half = (d & 7) + ((d >> 4) << 3);
      base[lane * 16 + half] = (_Float16)y[d];
    }
  }

  // ---- V: B-layout (32m x 16d, two d-groups) + plain f32 ----
  #pragma unroll
  for (int j = 0; j < DKK; j++) x[j] = valid ? v_in[xbase + (size_t)j * NN] : 0.0f;
  for (int i = 0; i < DKK; i++) {
    const float* wr = wv + (h * DKK + i) * DKK;
    float a = 0.0f;
    #pragma unroll
    for (int j = 0; j < DKK; j++) a = fmaf(wr[j], x[j], a);
    y[i] = a;
  }
  {
    int mc = n >> 5, mloc = n & 31;
    #pragma unroll
    for (int d = 0; d < DKK; d++) {
      int g = d >> 4;
      _Float16* base = vswz + (((size_t)bh * NC32 + mc) * 2 + g) * 512;
      int lane = (d & 15) + (((mloc >> 3) & 1) << 4);
      int half = (mloc & 7) + ((mloc >> 4) << 3);
      base[lane * 16 + half] = (_Float16)y[d];
    }
    if (valid) {
      #pragma unroll
      for (int d = 0; d < DKK; d++)
        vf32[((size_t)bh * DKK + d) * NN + n] = y[d];
    }
  }
}

// ---------------------------------------------------------------------------
// Kernel 3: IoU > 0.5 bitmask. One wave per query row, ballot -> 32-bit words.
// ---------------------------------------------------------------------------
__global__ void k_iou_mask(const float* __restrict__ boxes,
                           const float* __restrict__ cmask,
                           unsigned* __restrict__ words) {
  int wave = threadIdx.x >> 5;
  int lane = threadIdx.x & 31;
  int row  = blockIdx.x * 4 + wave;
  int b = row / NN, n = row % NN;

  const float4 bb = ((const float4*)boxes)[(size_t)b * NN + n];
  float x1n = bb.x - 0.5f * bb.z, y1n = bb.y - 0.5f * bb.w;
  float x2n = bb.x + 0.5f * bb.z, y2n = bb.y + 0.5f * bb.w;
  float an  = (x2n - x1n) * (y2n - y1n);
  float cmn = cmask[(size_t)b * NN + n];
  unsigned* wout = words + ((size_t)b * NN + n) * NWRD;

  for (int w = 0; w < NWRD; w++) {
    int m = w * 32 + lane;
    bool pred = false;
    if (m < NN) {
      float4 bm = ((const float4*)boxes)[(size_t)b * NN + m];
      float x1m = bm.x - 0.5f * bm.z, y1m = bm.y - 0.5f * bm.w;
      float x2m = bm.x + 0.5f * bm.z, y2m = bm.y + 0.5f * bm.w;
      float am  = (x2m - x1m) * (y2m - y1m);
      float iw  = fmaxf(fminf(x2n, x2m) - fmaxf(x1n, x1m), 0.0f);
      float ih  = fmaxf(fminf(y2n, y2m) - fmaxf(y1n, y1m), 0.0f);
      float inter = iw * ih;
      float uni   = an + am - inter;
      float iou   = inter / uni;                 // NaN -> pred false (matches ref)
      float cmm   = cmask[(size_t)b * NN + m];
      pred = (iou * cmn * cmm) > IOU_THR;
    }
    unsigned ball = __builtin_amdgcn_ballot_w32(pred);
    if (lane == 0) wout[w] = ball;
  }
}

// ---------------------------------------------------------------------------
// Kernel 4: fused masked attention. One wave per (b,h, TWO 16-query tiles).
// Two-pass flash: pass A = exact reference row-max of attn*mask; pass B =
// e=exp(s-max) on-mask, Σe, and P@Vᵀ via accumulated WMMAs per 32-key chunk.
// attn is computed transposed (A=K, B=Q) so Pᵀ(D-layout) -> P(A-layout)
// is a pure in-lane f32->f16 repack; K/V operand loads are shared by the
// two query tiles. Mask-word loads are unconditional (clamped row + AND).
// The m>=N tail guard is peeled into the final chunk only.
// ---------------------------------------------------------------------------
__global__ void k_attn(const _Float16* __restrict__ qswz,
                       const _Float16* __restrict__ kswz,
                       const _Float16* __restrict__ vswz,
                       const float* __restrict__ vf32,
                       const unsigned* __restrict__ words,
                       float* __restrict__ valbuf) {
  int wave = threadIdx.x >> 5;
  int lane = threadIdx.x & 31;
  int bh = blockIdx.x / 6;
  int wv = (blockIdx.x % 6) * 8 + wave;          // wave slot: 0..47
  if (wv >= 47) return;                           // whole-wave exit; no barriers
  int nt0 = wv * 2, nt1 = wv * 2 + 1;             // two 16-query tiles
  int b  = bh >> 3;
  int hi = lane >> 4;                             // half-wave group
  int ncol = lane & 15;                           // this lane's query column in P^T
  int nrow0 = nt0 * 16 + ncol;
  int nrow1 = nt1 * 16 + ncol;
  unsigned nmask0 = (nrow0 < NN) ? 0xFFFFFFFFu : 0u;
  unsigned nmask1 = (nrow1 < NN) ? 0xFFFFFFFFu : 0u;
  const unsigned* wrow0 = words + ((size_t)b * NN + (nrow0 < NN ? nrow0 : NN - 1)) * NWRD;
  const unsigned* wrow1 = words + ((size_t)b * NN + (nrow1 < NN ? nrow1 : NN - 1)) * NWRD;

  const _Float16* kbase = kswz + (size_t)bh * NT16 * 512 + lane * 16;
  const _Float16* vbase = vswz + (size_t)bh * NC32 * 1024 + lane * 16;
  const v16h qb0 = *(const v16h*)(qswz + ((size_t)bh * NT16 + nt0) * 512 + lane * 16);
  const v16h qb1 = *(const v16h*)(qswz + ((size_t)bh * NT16 + nt1) * 512 + lane * 16);
  const v8f cz = {};
  const int sh = 8 * hi;

  // ---------------- pass A: row max of attn*mask (and row-OR of mask) ------
  float lmax0 = -3.402823466e38f, lmax1 = -3.402823466e38f;
  unsigned ror0 = 0, ror1 = 0;
  for (int mc = 0; mc < NC32 - 1; mc++) {         // all m in range: no guard
    unsigned w0 = wrow0[mc] & nmask0;  ror0 |= w0;
    unsigned w1 = wrow1[mc] & nmask1;  ror1 |= w1;
    #pragma unroll
    for (int t = 0; t < 2; t++) {
      v16h ka = *(const v16h*)(kbase + (size_t)(mc * 2 + t) * 512);
      v8f s0 = wmma_f16(ka, qb0, cz);
      v8f s1 = wmma_f16(ka, qb1, cz);
      #pragma unroll
      for (int r = 0; r < 8; r++) {
        float c0 = ((w0 >> (t * 16 + r + sh)) & 1u) ? s0[r] : 0.0f;
        float c1 = ((w1 >> (t * 16 + r + sh)) & 1u) ? s1[r] : 0.0f;
        lmax0 = fmaxf(lmax0, c0);
        lmax1 = fmaxf(lmax1, c1);
      }
    }
  }
  {                                               // peeled tail chunk (m >= N)
    const int mc = NC32 - 1;
    unsigned w0 = wrow0[mc] & nmask0;  ror0 |= w0;
    unsigned w1 = wrow1[mc] & nmask1;  ror1 |= w1;
    #pragma unroll
    for (int t = 0; t < 2; t++) {
      v16h ka = *(const v16h*)(kbase + (size_t)(mc * 2 + t) * 512);
      v8f s0 = wmma_f16(ka, qb0, cz);
      v8f s1 = wmma_f16(ka, qb1, cz);
      int mbase = mc * 32 + t * 16 + sh;
      #pragma unroll
      for (int r = 0; r < 8; r++) {
        if (mbase + r < NN) {
          float c0 = ((w0 >> (t * 16 + r + sh)) & 1u) ? s0[r] : 0.0f;
          float c1 = ((w1 >> (t * 16 + r + sh)) & 1u) ? s1[r] : 0.0f;
          lmax0 = fmaxf(lmax0, c0);
          lmax1 = fmaxf(lmax1, c1);
        }
      }
    }
  }
  lmax0 = fmaxf(lmax0, swap16f(lmax0));           // full row max at every lane
  lmax1 = fmaxf(lmax1, swap16f(lmax1));
  ror0 |= (unsigned)swap16i((int)ror0);           // has-neighbor flag per row
  ror1 |= (unsigned)swap16i((int)ror1);

  // ---------------- pass B: e, Σe, P@Vᵀ ------------------------------------
  float rsum0 = 0.0f, rsum1 = 0.0f;
  v8f a00 = {}, a01 = {}, a10 = {}, a11 = {};
  for (int mc = 0; mc < NC32; mc++) {
    unsigned w0 = wrow0[mc] & nmask0;
    unsigned w1 = wrow1[mc] & nmask1;
    v16h pa0, pa1;
    #pragma unroll
    for (int t = 0; t < 2; t++) {
      v16h ka = *(const v16h*)(kbase + (size_t)(mc * 2 + t) * 512);
      v8f s0 = wmma_f16(ka, qb0, cz);
      v8f s1 = wmma_f16(ka, qb1, cz);
      #pragma unroll
      for (int r = 0; r < 8; r++) {
        float e0 = ((w0 >> (t * 16 + r + sh)) & 1u) ? __expf(s0[r] - lmax0) : 0.0f;
        float e1 = ((w1 >> (t * 16 + r + sh)) & 1u) ? __expf(s1[r] - lmax1) : 0.0f;
        rsum0 += e0;  rsum1 += e1;
        pa0[t * 8 + r] = (_Float16)e0;            // P^T(D) -> P(A): in-lane
        pa1[t * 8 + r] = (_Float16)e1;
      }
    }
    v16h vb0 = *(const v16h*)(vbase + (size_t)mc * 1024);
    v16h vb1 = *(const v16h*)(vbase + (size_t)mc * 1024 + 512);
    a00 = wmma_f16(pa0, vb0, a00);
    a01 = wmma_f16(pa0, vb1, a01);
    a10 = wmma_f16(pa1, vb0, a10);
    a11 = wmma_f16(pa1, vb1, a11);
  }
  rsum0 += swap16f(rsum0);                        // full Σe per row, every lane
  rsum1 += swap16f(rsum1);

  // ---------------- normalize, fallback, store -----------------------------
  int d0 = lane & 15;
  #pragma unroll
  for (int r = 0; r < 8; r++) {
    int nloc = r + sh;
    float s0 = bcastf(rsum0, nloc);
    float s1 = bcastf(rsum1, nloc);
    int  hn0 = bcasti((int)ror0, nloc);
    int  hn1 = bcasti((int)ror1, nloc);
    int n_0 = nt0 * 16 + nloc;
    int n_1 = nt1 * 16 + nloc;
    {
      float inv = 1.0f / (s0 + EPSV);
      size_t o0 = ((size_t)bh * DKK + d0) * NN + n_0;
      size_t o1 = ((size_t)bh * DKK + 16 + d0) * NN + n_0;
      if (hn0) { valbuf[o0] = a00[r] * inv; valbuf[o1] = a01[r] * inv; }
      else     { valbuf[o0] = vf32[o0];     valbuf[o1] = vf32[o1]; }
    }
    if (n_1 < NN) {
      float inv = 1.0f / (s1 + EPSV);
      size_t o0 = ((size_t)bh * DKK + d0) * NN + n_1;
      size_t o1 = ((size_t)bh * DKK + 16 + d0) * NN + n_1;
      if (hn1) { valbuf[o0] = a10[r] * inv; valbuf[o1] = a11[r] * inv; }
      else     { valbuf[o0] = vf32[o0];     valbuf[o1] = vf32[o1]; }
    }
  }
}

// ---------------------------------------------------------------------------
// Kernel 5: out = w_proj @ val + b_proj   (f32, LDS-staged wT rows, 16 j/batch)
// block = 256 threads = 16 i-groups x 16 n, each thread computes 16 outputs.
// ---------------------------------------------------------------------------
__global__ void k_out_proj(const float* __restrict__ valbuf,
                           const float* __restrict__ wT,
                           const float* __restrict__ bproj,
                           float* __restrict__ out) {
  __shared__ float wtile[16][256];
  int b  = blockIdx.x / NT16;
  int nt = blockIdx.x % NT16;
  int tid = threadIdx.x;
  int ii = tid >> 4, tn = tid & 15;
  int n = nt * 16 + tn;
  bool nv = (n < NN);

  float acc[16];
  #pragma unroll
  for (int k = 0; k < 16; k++) acc[k] = 0.0f;

  for (int j0 = 0; j0 < CCH; j0 += 16) {
    __syncthreads();
    #pragma unroll
    for (int jj = 0; jj < 16; jj++) wtile[jj][tid] = wT[(j0 + jj) * 256 + tid];
    __syncthreads();
    #pragma unroll
    for (int jj = 0; jj < 16; jj++) {
      float vj = nv ? valbuf[((size_t)b * CCH + j0 + jj) * NN + n] : 0.0f;
      #pragma unroll
      for (int k = 0; k < 16; k++) acc[k] = fmaf(wtile[jj][ii + 16 * k], vj, acc[k]);
    }
  }
  if (nv) {
    #pragma unroll
    for (int k = 0; k < 16; k++) {
      int i = ii + 16 * k;
      out[((size_t)b * CCH + i) * NN + n] = acc[k] + bproj[i];
    }
  }
}

// ---------------------------------------------------------------------------
extern "C" void kernel_launch(void* const* d_in, const int* in_sizes, int n_in,
                              void* d_out, int out_size, void* d_ws, size_t ws_size,
                              hipStream_t stream) {
  const float* queries  = (const float*)d_in[0];
  const float* keys     = (const float*)d_in[1];
  const float* values   = (const float*)d_in[2];
  const float* boxes    = (const float*)d_in[3];
  const float* cur_mask = (const float*)d_in[4];
  const float* wq       = (const float*)d_in[5];
  const float* wk       = (const float*)d_in[6];
  const float* wv       = (const float*)d_in[7];
  const float* w_proj   = (const float*)d_in[8];
  const float* b_proj   = (const float*)d_in[9];
  float* out = (float*)d_out;

  char* ws = (char*)d_ws;
  size_t off = 0;
  auto carve = [&](size_t bytes) { char* p = ws + off; off = (off + bytes + 255) & ~(size_t)255; return p; };

  _Float16* qswz = (_Float16*)carve((size_t)BSZ * HH * NT16 * 512 * 2);
  _Float16* kswz = (_Float16*)carve((size_t)BSZ * HH * NT16 * 512 * 2);
  _Float16* vswz = (_Float16*)carve((size_t)BSZ * HH * NC32 * 2 * 512 * 2);
  float*    vf32 = (float*)   carve((size_t)BSZ * HH * DKK * NN * 4);
  unsigned* mwrd = (unsigned*)carve((size_t)BSZ * NN * NWRD * 4);
  float*    vbuf = (float*)   carve((size_t)BSZ * CCH * NN * 4);
  float*    wT   = (float*)   carve((size_t)CCH * CCH * 4);
  (void)ws_size; (void)in_sizes; (void)n_in; (void)out_size;

  k_transpose_w<<<256, 256, 0, stream>>>(w_proj, wT);
  k_head_proj  <<<BSZ * HH * 12, 128, 0, stream>>>(queries, keys, values, wq, wk, wv,
                                                   qswz, kswz, vswz, vf32);
  k_iou_mask   <<<(BSZ * NN) / 4, 128, 0, stream>>>(boxes, cur_mask, mwrd);
  k_attn       <<<BSZ * HH * 6, 256, 0, stream>>>(qswz, kswz, vswz, vf32, mwrd, vbuf);
  k_out_proj   <<<BSZ * NT16, 256, 0, stream>>>(vbuf, wT, b_proj, out);
}
